// RelativeMultiHeadDotProductAttention_85787676770422
// MI455X (gfx1250) — matrix operand order repaired and checked by reading, not compiled, not run on hardware
//
#include <hip/hip_runtime.h>
#include <hip/hip_bf16.h>

typedef __bf16 bf16;
typedef __attribute__((ext_vector_type(16))) __bf16 v16bf;
typedef __attribute__((ext_vector_type(8)))  __bf16 v8bf;
typedef __attribute__((ext_vector_type(4)))  __bf16 v4bf;
typedef __attribute__((ext_vector_type(8)))  float  v8f;
typedef __attribute__((ext_vector_type(4)))  float  v4f;

#define DEV static __device__ __forceinline__

DEV v16bf cat8(v8bf lo, v8bf hi) {
  return __builtin_shufflevector(lo, hi, 0,1,2,3,4,5,6,7,8,9,10,11,12,13,14,15);
}

// A-fragment (16x32 bf16, M x K). lane&15 selects the M row (caller bakes it into
// rowPtr); half = lane>>4. Elements 0-7 hold K = kb + half*8 + e, elements 8-15
// hold K = kb + half*8 + 16 + e  (per ISA 16-bit A-matrix table).
DEV v16bf load_afrag(const bf16* rowPtr, int kb, int half) {
  const bf16* p = rowPtr + kb + half * 8;
  v8bf lo = *(const v8bf*)(p);
  v8bf hi = *(const v8bf*)(p + 16);
  return cat8(lo, hi);
}

// B-fragment (32x16 bf16, K x N) read from a K-contiguous row (Bt layout):
// lane&15 selects N (caller bakes into rowPtr); elements 0-7 hold K = kb + half*8 + e,
// elements 8-15 hold K = kb + half*8 + 16 + e  (same K order as load_afrag).
DEV v16bf load_bfrag(const bf16* rowPtr, int kb, int half) {
  const bf16* p = rowPtr + kb + half * 8;
  v8bf lo = *(const v8bf*)(p);
  v8bf hi = *(const v8bf*)(p + 16);
  return cat8(lo, hi);
}

DEV v8f wmma_bf16(v16bf a, v16bf b, v8f c) {
  return __builtin_amdgcn_wmma_f32_16x16x32_bf16(false, a, false, b, (short)0, c,
                                                 false, false);
}

// ---------------------------------------------------------------- conversions
__global__ __launch_bounds__(256) void cvt_f32_bf16(const float* __restrict__ in,
                                                    bf16* __restrict__ out, int n) {
  int i = (blockIdx.x * blockDim.x + threadIdx.x) * 4;
  if (i + 3 < n) {
    v4f x = *(const v4f*)(in + i);
    v4bf o = { (bf16)x[0], (bf16)x[1], (bf16)x[2], (bf16)x[3] };
    *(v4bf*)(out + i) = o;
  }
}

// out[c*rows + r] = (bf16) in[r*cols + c]   (weights -> Bt layout, K-contiguous)
__global__ __launch_bounds__(256) void transpose_w(const float* __restrict__ in,
                                                   bf16* __restrict__ out,
                                                   int rows, int cols) {
  int idx = blockIdx.x * blockDim.x + threadIdx.x;
  if (idx < rows * cols) {
    int r = idx / cols, c = idx % cols;
    out[c * rows + r] = (bf16)in[idx];
  }
}

// ---------------------------------------------------------------------- GEMM
// C = A[MxK](bf16,row-major) * Bt[NxK](bf16)  + bias[n], scaled.
// MODE 0: fp32 output, row-major [M x N]
// MODE 1: bf16 output scattered to [B, H, S, HD]  (m = b*S+s, n = h*HD+hd)
// LDS double-buffered: stage tile k+1 while computing tile k (1 barrier/iter).
template <int MODE>
__global__ __launch_bounds__(256) void gemm_kernel(
    const bf16* __restrict__ A, const bf16* __restrict__ Bt,
    const float* __restrict__ bias, float scale,
    float* __restrict__ outF, bf16* __restrict__ outB,
    int M, int N, int K, int Sdim, int Hn, int HDdim) {
  const int LSTR = 48;           // 32 + 16 pad; 96B row stride (16B aligned)
  const int BUF  = 128 * LSTR;   // one tile
  __shared__ __align__(16) bf16 sA[2 * BUF];
  __shared__ __align__(16) bf16 sB[2 * BUF];

  int tid  = threadIdx.x;
  int lane = tid & 31, wave = tid >> 5;
  int wm = wave >> 1, wn = wave & 1;  // 4 x 2 wave grid
  int n15 = lane & 15, half = lane >> 4;
  int mBase = blockIdx.x * 128;
  int nBase = blockIdx.y * 128;

  // per-thread staging coordinates (2 x v8bf per tile per array)
  int sRow0 = tid >> 2, sKg0 = (tid & 3) * 8;
  int sRow1 = (tid + 256) >> 2, sKg1 = ((tid + 256) & 3) * 8;
  int aOff0 = (mBase + sRow0) * K + sKg0, aOff1 = (mBase + sRow1) * K + sKg1;
  int bOff0 = (nBase + sRow0) * K + sKg0, bOff1 = (nBase + sRow1) * K + sKg1;

  v8f acc[2][4] = {};

  // stage tile 0 into buffer 0
  *(v8bf*)(sA + sRow0 * LSTR + sKg0) = *(const v8bf*)(A + aOff0);
  *(v8bf*)(sA + sRow1 * LSTR + sKg1) = *(const v8bf*)(A + aOff1);
  *(v8bf*)(sB + sRow0 * LSTR + sKg0) = *(const v8bf*)(Bt + bOff0);
  *(v8bf*)(sB + sRow1 * LSTR + sKg1) = *(const v8bf*)(Bt + bOff1);
  __syncthreads();

  for (int k0 = 0; k0 < K; k0 += 32) {
    int cur = (k0 >> 5) & 1;
    int nxt = cur ^ 1;
    // issue next tile's global->LDS staging first (overlaps with WMMA below)
    if (k0 + 32 < K) {
      int kn = k0 + 32;
      *(v8bf*)(sA + nxt * BUF + sRow0 * LSTR + sKg0) = *(const v8bf*)(A + aOff0 + kn);
      *(v8bf*)(sA + nxt * BUF + sRow1 * LSTR + sKg1) = *(const v8bf*)(A + aOff1 + kn);
      *(v8bf*)(sB + nxt * BUF + sRow0 * LSTR + sKg0) = *(const v8bf*)(Bt + bOff0 + kn);
      *(v8bf*)(sB + nxt * BUF + sRow1 * LSTR + sKg1) = *(const v8bf*)(Bt + bOff1 + kn);
    }
    const bf16* cA = sA + cur * BUF;
    const bf16* cB = sB + cur * BUF;
    v16bf afr[2];
#pragma unroll
    for (int mt = 0; mt < 2; ++mt)
      afr[mt] = load_afrag(cA + (wm * 32 + mt * 16 + n15) * LSTR, 0, half);
#pragma unroll
    for (int nt = 0; nt < 4; ++nt) {
      v16bf bfr = load_bfrag(cB + (wn * 64 + nt * 16 + n15) * LSTR, 0, half);
#pragma unroll
      for (int mt = 0; mt < 2; ++mt)
        acc[mt][nt] = wmma_bf16(afr[mt], bfr, acc[mt][nt]);
    }
    __syncthreads();  // next iter reads buf[nxt]; this iter's reads/writes done
  }

#pragma unroll
  for (int mt = 0; mt < 2; ++mt) {
#pragma unroll
    for (int nt = 0; nt < 4; ++nt) {
      int n = nBase + wn * 64 + nt * 16 + n15;
      float bv = bias[n];
#pragma unroll
      for (int r = 0; r < 8; ++r) {
        int m = mBase + wm * 32 + mt * 16 + r + 8 * half;
        float v = (acc[mt][nt][r] + bv) * scale;
        if (MODE == 0) {
          outF[m * N + n] = v;
        } else {
          int b = m >> 10, s = m & 1023;  // S = 1024
          int h = n >> 6, hd = n & 63;    // HD = 64
          outB[((b * Hn + h) * Sdim + s) * HDdim + hd] = (bf16)v;
        }
      }
    }
  }
}

// ----------------------------------------------------------- flash attention
// grid: (S/64, B*H), block: 128 threads (4 waves, 16 query rows per wave)
__global__ __launch_bounds__(128) void attn_kernel(
    const bf16* __restrict__ Q, const bf16* __restrict__ Kb,
    const bf16* __restrict__ Vb, const float* __restrict__ rel_bias,
    const int* __restrict__ pos, bf16* __restrict__ X, int S, int Hn) {
  const int KSTR = 80;  // 64 + 16 pad; 160B row stride
  __shared__ __align__(16) bf16 sK[64 * KSTR];      // [key][hd]
  __shared__ __align__(16) bf16 sVt[64 * KSTR];     // [hd][key]
  __shared__ __align__(16) bf16 sP[4 * 16 * KSTR];  // per-wave [16 q][64 key]

  int tid  = threadIdx.x;
  int lane = tid & 31, wave = tid >> 5;
  int n15 = lane & 15, half = lane >> 4;
  int bh = blockIdx.y;
  int h  = bh % Hn;
  int q0 = blockIdx.x * 64 + wave * 16;
  const bf16* Qb = Q + bh * S * 64;
  const bf16* Kp = Kb + bh * S * 64;
  const bf16* Vp = Vb + bh * S * 64;

  // Q fragments stay in registers for the whole kernel (HD=64 -> 2 k-subtiles)
  v16bf aq[2];
#pragma unroll
  for (int ks = 0; ks < 2; ++ks)
    aq[ks] = load_afrag(Qb + (q0 + n15) * 64, ks * 32, half);

  v8f o[4] = {};
  float mrow[8], lrow[8];
#pragma unroll
  for (int r = 0; r < 8; ++r) { mrow[r] = -3.0e38f; lrow[r] = 0.f; }

  bf16* myP = sP + wave * 16 * KSTR;

  for (int c0 = 0; c0 < S; c0 += 64) {
    __syncthreads();
    // stage K chunk [64][64] and V^T [64 hd][64 key]; prefetch next chunk
#pragma unroll
    for (int it = 0; it < 4; ++it) {
      int idx = tid + it * 128;
      int row = idx >> 3;
      int cg  = (idx & 7) * 8;
      *(v8bf*)(sK + row * KSTR + cg) = *(const v8bf*)(Kp + (c0 + row) * 64 + cg);
      v8bf vv = *(const v8bf*)(Vp + (c0 + row) * 64 + cg);
#pragma unroll
      for (int e = 0; e < 8; ++e) sVt[(cg + e) * KSTR + row] = vv[e];
      if (c0 + 64 < S) {  // speculative prefetch of next K/V chunk
        __builtin_prefetch(Kp + (c0 + 64 + row) * 64 + cg, 0, 0);
        __builtin_prefetch(Vp + (c0 + 64 + row) * 64 + cg, 0, 0);
      }
    }
    __syncthreads();

    // scores: 16 queries x 64 keys via WMMA (K^T rows are K-contiguous)
    v8f sc[4];
#pragma unroll
    for (int nt = 0; nt < 4; ++nt) {
      v8f c = {};
#pragma unroll
      for (int ks = 0; ks < 2; ++ks) {
        v16bf bk = load_bfrag(sK + (nt * 16 + n15) * KSTR, ks * 32, half);
        c = wmma_bf16(aq[ks], bk, c);
      }
      sc[nt] = c;
    }

    // relative-position bias gather (pure int32 addressing) + row max
    float rmax[8];
#pragma unroll
    for (int r = 0; r < 8; ++r) rmax[r] = -3.0e38f;
#pragma unroll
    for (int nt = 0; nt < 4; ++nt) {
      int key = c0 + nt * 16 + n15;
#pragma unroll
      for (int r = 0; r < 8; ++r) {
        int q = q0 + r + 8 * half;
        int pidx = pos[q * S + key];
        float v = sc[nt][r] + rel_bias[pidx * Hn + h];
        sc[nt][r] = v;
        rmax[r] = fmaxf(rmax[r], v);
      }
    }
#pragma unroll
    for (int r = 0; r < 8; ++r)
#pragma unroll
      for (int off = 1; off < 16; off <<= 1)
        rmax[r] = fmaxf(rmax[r], __shfl_xor(rmax[r], off, 32));

    // online softmax update
#pragma unroll
    for (int r = 0; r < 8; ++r) {
      float mn = fmaxf(mrow[r], rmax[r]);
      float al = __expf(mrow[r] - mn);
      float part = 0.f;
#pragma unroll
      for (int nt = 0; nt < 4; ++nt) {
        float p = __expf(sc[nt][r] - mn);
        sc[nt][r] = p;
        part += p;
      }
#pragma unroll
      for (int off = 1; off < 16; off <<= 1)
        part += __shfl_xor(part, off, 32);
      lrow[r] = lrow[r] * al + part;
      mrow[r] = mn;
#pragma unroll
      for (int ht = 0; ht < 4; ++ht) o[ht][r] *= al;
    }

    // P -> per-wave LDS (bf16, row-major); same-wave LDS ops are in-order
#pragma unroll
    for (int nt = 0; nt < 4; ++nt)
#pragma unroll
      for (int r = 0; r < 8; ++r)
        myP[(r + 8 * half) * KSTR + nt * 16 + n15] = (bf16)sc[nt][r];

    // O += P @ V   (A = P rows, B = V^T rows, both K-contiguous)
#pragma unroll
    for (int ks = 0; ks < 2; ++ks) {
      v16bf ap = load_afrag(myP + n15 * KSTR, ks * 32, half);
#pragma unroll
      for (int ht = 0; ht < 4; ++ht) {
        v16bf bv = load_bfrag(sVt + (ht * 16 + n15) * KSTR, ks * 32, half);
        o[ht] = wmma_bf16(ap, bv, o[ht]);
      }
    }
  }

  // normalize, store X[b*S+q][h*64+hd] in bf16 for the output GEMM
  int b = bh / Hn;
  int D = Hn * 64;
#pragma unroll
  for (int ht = 0; ht < 4; ++ht)
#pragma unroll
    for (int r = 0; r < 8; ++r) {
      int q = q0 + r + 8 * half;
      float v = o[ht][r] / lrow[r];
      X[(b * S + q) * D + h * 64 + ht * 16 + n15] = (bf16)v;
    }
}

// -------------------------------------------------------------------- launch
extern "C" void kernel_launch(void* const* d_in, const int* in_sizes, int n_in,
                              void* d_out, int out_size, void* d_ws, size_t ws_size,
                              hipStream_t stream) {
  (void)in_sizes; (void)n_in; (void)out_size; (void)ws_size;
  const int B = 8, S = 1024, D = 768, H = 12, HD = 64;
  const int M = B * S;  // 8192

  const float* inq  = (const float*)d_in[0];
  const float* inkv = (const float*)d_in[1];
  const float* Wq   = (const float*)d_in[2];
  const float* bq   = (const float*)d_in[3];
  const float* Wk   = (const float*)d_in[4];
  const float* bk   = (const float*)d_in[5];
  const float* Wv   = (const float*)d_in[6];
  const float* bv   = (const float*)d_in[7];
  const float* Wo   = (const float*)d_in[8];
  const float* bo   = (const float*)d_in[9];
  const float* rel  = (const float*)d_in[10];
  const int*   pos  = (const int*)d_in[11];
  float* out = (float*)d_out;

  char* w = (char*)d_ws;
  auto carve = [&](size_t bytes) -> char* {
    char* p = w;
    w += (bytes + 255) & ~(size_t)255;
    return p;
  };
  bf16* Xq   = (bf16*)carve((size_t)M * D * 2);
  bf16* Xkv  = (bf16*)carve((size_t)M * D * 2);
  bf16* Wqt  = (bf16*)carve((size_t)D * D * 2);
  bf16* Wkt  = (bf16*)carve((size_t)D * D * 2);
  bf16* Wvt  = (bf16*)carve((size_t)D * D * 2);
  bf16* Wot  = (bf16*)carve((size_t)D * D * 2);
  bf16* Qb   = (bf16*)carve((size_t)M * D * 2);  // [B,H,S,HD]
  bf16* Kbuf = (bf16*)carve((size_t)M * D * 2);
  bf16* Vbuf = (bf16*)carve((size_t)M * D * 2);
  bf16* Xa   = (bf16*)carve((size_t)M * D * 2);  // attention output [M, D]

  // 1) activations fp32 -> bf16; weights fp32 -> transposed bf16 (Bt layout)
  cvt_f32_bf16<<<(M * D / 4) / 256, 256, 0, stream>>>(inq, Xq, M * D);
  cvt_f32_bf16<<<(M * D / 4) / 256, 256, 0, stream>>>(inkv, Xkv, M * D);
  int tw = (D * D + 255) / 256;
  transpose_w<<<tw, 256, 0, stream>>>(Wq, Wqt, D, D);
  transpose_w<<<tw, 256, 0, stream>>>(Wk, Wkt, D, D);
  transpose_w<<<tw, 256, 0, stream>>>(Wv, Wvt, D, D);
  transpose_w<<<tw, 256, 0, stream>>>(Wo, Wot, D, D);

  // 2) QKV projections (Q pre-scaled by 1/sqrt(HD)), scattered to [B,H,S,HD]
  dim3 gg(M / 128, D / 128);
  gemm_kernel<1><<<gg, 256, 0, stream>>>(Xq, Wqt, bq, 0.125f, nullptr, Qb,
                                         M, D, D, S, H, HD);
  gemm_kernel<1><<<gg, 256, 0, stream>>>(Xkv, Wkt, bk, 1.0f, nullptr, Kbuf,
                                         M, D, D, S, H, HD);
  gemm_kernel<1><<<gg, 256, 0, stream>>>(Xkv, Wvt, bv, 1.0f, nullptr, Vbuf,
                                         M, D, D, S, H, HD);

  // 3) fused bias + flash softmax + PV
  attn_kernel<<<dim3(S / 64, B * H), 128, 0, stream>>>(Qb, Kbuf, Vbuf, rel, pos,
                                                       Xa, S, H);

  // 4) output projection -> fp32 d_out
  gemm_kernel<0><<<gg, 256, 0, stream>>>(Xa, Wot, bo, 1.0f, out, nullptr,
                                         M, D, D, S, H, HD);
}
